// SNNExoplanetDetector_62139586838605
// MI455X (gfx1250) — matrix operand order repaired and checked by reading, not compiled, not run on hardware
//
#include <hip/hip_runtime.h>

typedef __attribute__((ext_vector_type(16))) __bf16 v16bf;
typedef __attribute__((ext_vector_type(8)))  __bf16 v8bf;
typedef __attribute__((ext_vector_type(4)))  __bf16 v4bf;
typedef __attribute__((ext_vector_type(8)))  float  v8f;

namespace {

constexpr int kB  = 256;
constexpr int kT  = 1000;
constexpr int kF  = 128;
constexpr int kH  = 256;
constexpr int kO  = 2;
constexpr int kBT = 16;        // batch rows per workgroup (WMMA M)
constexpr int kThreads = 512;  // 16 wave32, one 16-col N-tile each

// LDS layout (bytes): double-buffered spikes + x_t staging + activity
constexpr int kZElems  = kBT * kH;                   // one z buffer (bf16)
constexpr int kXtElems = kBT * kF;                   // one xt buffer (bf16)
constexpr int kZOff    = 0;                          // 2 * 8192 B = 16384
constexpr int kXtOff   = kZOff  + 2 * kZElems * 2;   // 2 * 4096 B = 8192
constexpr int kActOff  = kXtOff + 2 * kXtElems * 2;  // 16*256*4   = 16384
constexpr int kLdsBytes = kActOff + kBT * kH * 4;    // 40960 total

__device__ inline v16bf cat16(v8bf lo, v8bf hi) {
  return __builtin_shufflevector(lo, hi, 0, 1, 2, 3, 4, 5, 6, 7,
                                         8, 9, 10, 11, 12, 13, 14, 15);
}

// A fragment: 16x32 bf16 tile from row-major [16][kstride] LDS buffer.
// ISA 16-bit A layout: lanes 0-15 hold K[k0..k0+7] & K[k0+16..k0+23],
// lanes 16-31 hold K[k0+8..k0+15] & K[k0+24..k0+31]; row = lane&15.
__device__ inline v16bf load_a(const __bf16* base, int kstride, int k0, int lane) {
  const int row = lane & 15;
  const int kb  = k0 + ((lane & 16) ? 8 : 0);
  const __bf16* p = base + row * kstride + kb;
  v8bf lo = *(const v8bf*)p;         // ds_load_b128
  v8bf hi = *(const v8bf*)(p + 16);  // ds_load_b128
  return cat16(lo, hi);
}

// B fragment: 32x16 bf16 tile built directly from GLOBAL fp32 weights stored
// [N][K] row-major (B column n = W row n). Lanes 0-15: K[k0..k0+7] & K[k0+16..k0+23], col=lane;
// lanes 16-31: K[k0+8..k0+15] & K[k0+24..k0+31], col=lane-16. Held in VGPRs for the whole scan.
__device__ inline v16bf load_b_frag(const float* w, int kstride, int nbase,
                                    int k0, int lane) {
  const int n  = nbase + (lane & 15);
  const int kb = k0 + ((lane & 16) ? 8 : 0);
  const float* p = w + n * kstride + kb;
  float4 f0 = ((const float4*)p)[0];
  float4 f1 = ((const float4*)p)[1];
  float4 f2 = ((const float4*)p)[4];
  float4 f3 = ((const float4*)p)[5];
  v16bf r = {(__bf16)f0.x, (__bf16)f0.y, (__bf16)f0.z, (__bf16)f0.w,
             (__bf16)f1.x, (__bf16)f1.y, (__bf16)f1.z, (__bf16)f1.w,
             (__bf16)f2.x, (__bf16)f2.y, (__bf16)f2.z, (__bf16)f2.w,
             (__bf16)f3.x, (__bf16)f3.y, (__bf16)f3.z, (__bf16)f3.w};
  return r;
}

__device__ inline v8f wmma_bf16(v16bf a, v16bf b, v8f c) {
  return __builtin_amdgcn_wmma_f32_16x16x32_bf16(false, a, false, b,
                                                 (short)0, c, false, false);
}

} // namespace

__global__ __launch_bounds__(kThreads)
void snn_lif_scan_kernel(const float* __restrict__ x,
                         const float* __restrict__ w_in,
                         const float* __restrict__ w_rec,
                         const float* __restrict__ fc_w,
                         const float* __restrict__ fc_b,
                         float* __restrict__ out) {
  extern __shared__ char smem[];
  __bf16* lds_zb  = (__bf16*)(smem + kZOff);    // [2][16][256]
  __bf16* lds_xtb = (__bf16*)(smem + kXtOff);   // [2][16][128]
  float*  lds_act = (float*)(smem + kActOff);   // [16][256]

  const int tid  = threadIdx.x;
  const int lane = tid & 31;
  const int wave = tid >> 5;       // 0..15
  const int b0   = blockIdx.x * kBT;
  const int nb   = wave * 16;      // this wave's N-tile base column

  // ---- hoist all weight B-fragments into VGPRs (loop-invariant) ----
  v16bf win_f[4], wrec_f[8];
#pragma unroll
  for (int ks = 0; ks < 4; ++ks)
    win_f[ks] = load_b_frag(w_in, kF, nb, ks * 32, lane);
#pragma unroll
  for (int ks = 0; ks < 8; ++ks)
    wrec_f[ks] = load_b_frag(w_rec, kH, nb, ks * 32, lane);

  // ---- prologue: zero z-buffer 0, stage x(0) into xt-buffer 0 ----
  {
    v8bf zv = {};
    ((v8bf*)lds_zb)[tid] = zv;  // 512 threads x 8 halfs = 16x256 (buffer 0)
  }
  const int xrow = tid >> 5;          // 0..15
  const int xcol = (tid & 31) * 4;    // 0,4,...,124
  const float* xbase = x + ((size_t)(b0 + xrow) * kT) * kF + xcol;
  {
    float4 g = *(const float4*)xbase;
    v4bf xv = {(__bf16)g.x, (__bf16)g.y, (__bf16)g.z, (__bf16)g.w};
    *(v4bf*)(lds_xtb + xrow * kF + xcol) = xv;  // buffer 0
  }
  __syncthreads();

  // LIF state (fp32 VGPRs): VGPR r <-> row r + (lane>=16 ? 8 : 0)
  v8f vm = {}, cur = {}, act = {};
  const int mb = (lane & 16) ? 8 : 0;
  const int nc = nb + (lane & 15);

  for (int t = 0; t < kT; ++t) {
    const int cb   = t & 1;      // buffers read this step
    const int nbuf = cb ^ 1;     // buffers written for next step
    const __bf16* z_rd  = lds_zb  + cb * kZElems;
    const __bf16* xt_rd = lds_xtb + cb * kXtElems;
    __bf16* z_wr  = lds_zb  + nbuf * kZElems;
    __bf16* xt_wr = lds_xtb + nbuf * kXtElems;

    // issue global loads for x(t+1) now; consume after the WMMA chain
    float4 g = {};
    if (t + 1 < kT) {
      g = *(const float4*)(xbase + (size_t)(t + 1) * kF);
      if (t + 2 < kT)  // -> global_prefetch_b8
        __builtin_prefetch(xbase + (size_t)(t + 2) * kF, 0, 0);
    }

    // acc = x_t @ W_in^T + z @ W_rec^T  (bf16 WMMA, fp32 accumulate)
    v8f acc = {};
#pragma unroll
    for (int ks = 0; ks < 4; ++ks) {
      v16bf a = load_a(xt_rd, kF, ks * 32, lane);
      acc = wmma_bf16(a, win_f[ks], acc);
    }
#pragma unroll
    for (int ks = 0; ks < 8; ++ks) {
      v16bf a = load_a(z_rd, kH, ks * 32, lane);
      acc = wmma_bf16(a, wrec_f[ks], acc);
    }

    // stage x(t+1) into the alternate buffer (no hazard: readers use cb)
    if (t + 1 < kT) {
      v4bf xv = {(__bf16)g.x, (__bf16)g.y, (__bf16)g.z, (__bf16)g.w};
      *(v4bf*)(xt_wr + xrow * kF + xcol) = xv;
    }

    // LIF update: v_dec = 0.9v + 0.1i ; i' = 0.8i + acc ; spike/reset
#pragma unroll
    for (int r = 0; r < 8; ++r) {
      float vd = 0.9f * vm[r] + 0.1f * cur[r];
      float zn = (vd > 1.0f) ? 1.0f : 0.0f;
      cur[r] = 0.8f * cur[r] + acc[r];
      vm[r]  = vd * (1.0f - zn);
      act[r] += zn;
      z_wr[(mb + r) * kH + nc] = (__bf16)zn;
    }

    __syncthreads();  // single barrier: next-step buffers now visible
  }

  // ---- epilogue: total_activity @ fc_w^T + fc_b  (O=2, tiny) ----
#pragma unroll
  for (int r = 0; r < 8; ++r)
    lds_act[(mb + r) * kH + nc] = act[r];
  __syncthreads();
  if (tid < kBT * kO) {
    const int bl = tid >> 1;
    const int o  = tid & 1;
    float s = fc_b[o];
    for (int h = 0; h < kH; ++h)
      s += lds_act[bl * kH + h] * fc_w[o * kH + h];
    out[(b0 + bl) * kO + o] = s;
  }
}

extern "C" void kernel_launch(void* const* d_in, const int* in_sizes, int n_in,
                              void* d_out, int out_size, void* d_ws, size_t ws_size,
                              hipStream_t stream) {
  (void)in_sizes; (void)n_in; (void)out_size; (void)d_ws; (void)ws_size;
  const float* x     = (const float*)d_in[0];
  const float* w_in  = (const float*)d_in[1];
  const float* w_rec = (const float*)d_in[2];
  const float* fc_w  = (const float*)d_in[3];
  const float* fc_b  = (const float*)d_in[4];
  float* out = (float*)d_out;

  snn_lif_scan_kernel<<<dim3(kB / kBT), dim3(kThreads), kLdsBytes, stream>>>(
      x, w_in, w_rec, fc_w, fc_b, out);
}
